// ContextProjector_32658931319370
// MI455X (gfx1250) — matrix operand order repaired and checked by reading, not compiled, not run on hardware
//
#include <hip/hip_runtime.h>

#define B     8
#define N     4096
#define MG    2048
#define KNN   32
#define BIGF  1e10f

typedef __attribute__((ext_vector_type(16))) _Float16 v16h;
typedef __attribute__((ext_vector_type(8)))  _Float16 v8h;
typedef __attribute__((ext_vector_type(8)))  float    v8f;

union HFrag { v16h v; v8h h8[2]; _Float16 h[16]; };
union FFrag { v8f  v; float    f[8]; };

// Branchless erf (Abramowitz-Stegun 7.1.26, |abs err| <= 1.5e-7).
// Uses hardware v_rcp_f32 + v_exp_f32 (error far below the f16 WMMA noise).
__device__ __forceinline__ float erf_fast(float z) {
  float a = fabsf(z);
  float t = __builtin_amdgcn_rcpf(fmaf(0.3275911f, a, 1.0f));
  float p = 1.061405429f;
  p = fmaf(p, t, -1.453152027f);
  p = fmaf(p, t, 1.421413741f);
  p = fmaf(p, t, -0.284496736f);
  p = fmaf(p, t, 0.254829592f);
  p *= t;
  float y = fmaf(-p, __expf(-a * a), 1.0f);
  return copysignf(y, z);
}

__device__ __forceinline__ float gelu_exact(float x) {
  return 0.5f * x * (1.0f + erf_fast(x * 0.70710678118654752f));
}

// ---------------------------------------------------------------------------
// Ball query: one thread per query, whole per-batch point set in LDS.
// Unsorted top-K (K smallest d2 within radius) with worst-slot tracking.
// Only the SET of indices matters downstream (masked mean is order-free).
// ---------------------------------------------------------------------------
__global__ void __launch_bounds__(64)
ballquery_kernel(const float* __restrict__ qpos, const float* __restrict__ ppos,
                 int Q, int P, float r2, int* __restrict__ idxOut) {
  __shared__ float spx[4096];
  __shared__ float spy[4096];
  __shared__ float sD[64 * KNN];
  __shared__ int   sI[64 * KNN];

  int nQB = (Q + 63) / 64;
  int b   = blockIdx.x / nQB;
  int qb  = blockIdx.x % nQB;
  int tid = threadIdx.x;

  const float2* pp2 = (const float2*)(ppos + (size_t)b * P * 2);
  for (int i = tid; i < P; i += 64) {
    float2 p = pp2[i];
    spx[i] = p.x;
    spy[i] = p.y;
  }
  __syncthreads();

  int q = qb * 64 + tid;
  if (q >= Q) return;

  float qx = qpos[((size_t)b * Q + q) * 2 + 0];
  float qy = qpos[((size_t)b * Q + q) * 2 + 1];

  int base = tid * KNN;
  for (int j = 0; j < KNN; ++j) { sD[base + j] = BIGF; sI[base + j] = -1; }

  float worstVal = BIGF;
  int   worstSlot = 0;
  for (int p = 0; p < P; ++p) {
    float dx = qx - spx[p];
    float dy = qy - spy[p];
    float d2 = dx * dx + dy * dy;
    if (d2 <= r2 && d2 < worstVal) {
      sD[base + worstSlot] = d2;
      sI[base + worstSlot] = p;
      float wv = -1.0f; int ws = 0;
      for (int j = 0; j < KNN; ++j) {
        float v = sD[base + j];
        if (v > wv) { wv = v; ws = j; }
      }
      worstVal = wv; worstSlot = ws;
    }
  }

  int4* o4 = (int4*)(idxOut + ((size_t)b * Q + q) * KNN);
  const int4* s4 = (const int4*)(sI + base);
  #pragma unroll
  for (int j = 0; j < KNN / 4; ++j) o4[j] = s4[j];
}

// ---------------------------------------------------------------------------
// Gathered 2-layer MLP + masked mean, one wave per query, WMMA f16->f32.
//   X (32 x DIN, zero-padded) -> gelu -> 64 -> 64, masked mean over K rows.
// Weight B-fragments preloaded into registers (query-invariant); X/H row-major
// in LDS so A-fragments are 2x ds_load_b128 each.
// MODE 0: write per-query mean (B,Q,64)  (phi path, reduced later)
// MODE 1: write augmentation (B,Q,128) at column offset colOff (psi path)
// ---------------------------------------------------------------------------
template<int DF, int DIN, int NKC1, int MODE>
__global__ void __launch_bounds__(128)
mlp_gather_kernel(const float* __restrict__ feats,
                  const float* __restrict__ qpos,
                  const float* __restrict__ ppos,
                  const int*   __restrict__ idxIn,
                  const float* __restrict__ w1, const float* __restrict__ b1,
                  const float* __restrict__ w2, const float* __restrict__ b2,
                  float* __restrict__ outP, int Q, int P, int colOff) {
  __shared__ _Float16 sW1t[64 * 64];   // transposed: [n][k]
  __shared__ _Float16 sW2t[64 * 64];   // transposed: [n][k]
  __shared__ float    sB1v[64];
  __shared__ float    sB2v[64];
  __shared__ _Float16 sX[4][32 * 64];
  __shared__ _Float16 sH[4][32 * 64];

  int tid = threadIdx.x;
  for (int i = tid; i < 64 * 64; i += 128) {
    int c = i >> 6, k = i & 63;
    sW1t[i] = (k < DIN) ? (_Float16)w1[k * 64 + c] : (_Float16)0.0f;
    sW2t[i] = (_Float16)w2[k * 64 + c];
  }
  if (tid < 64) { sB1v[tid] = b1[tid]; sB2v[tid] = b2[tid]; }
  __syncthreads();

  const int w     = tid >> 5;
  const int lane  = tid & 31;
  const int half8 = (lane & 16) ? 8 : 0;   // ISA 7.12.2 wave32 layouts
  const int l15   = lane & 15;
  const int klo   = (lane & 16) ? 16 : 0;
  _Float16* xw = sX[w];
  _Float16* hw = sH[w];

  // Preload query-invariant weight B-fragments into registers (2x b128 each).
  HFrag bf1[NKC1][4];
  #pragma unroll
  for (int kc = 0; kc < NKC1; ++kc) {
    #pragma unroll
    for (int nt = 0; nt < 4; ++nt) {
      const _Float16* src = &sW1t[(nt * 16 + l15) * 64 + kc * 32 + klo];
      bf1[kc][nt].h8[0] = *(const v8h*)(src);
      bf1[kc][nt].h8[1] = *(const v8h*)(src + 8);
    }
  }
  HFrag bf2[2][4];
  #pragma unroll
  for (int kc = 0; kc < 2; ++kc) {
    #pragma unroll
    for (int nt = 0; nt < 4; ++nt) {
      const _Float16* src = &sW2t[(nt * 16 + l15) * 64 + kc * 32 + klo];
      bf2[kc][nt].h8[0] = *(const v8h*)(src);
      bf2[kc][nt].h8[1] = *(const v8h*)(src + 8);
    }
  }

  const long long BQ = (long long)B * Q;
  const long long stride = (long long)gridDim.x * 4;

  for (long long wid = (long long)blockIdx.x * 4 + w; wid < BQ; wid += stride) {
    int b = (int)(wid / Q);
    int q = (int)(wid % Q);

    // ---- stage this lane's gathered row (row index == lane) ----
    int myIdx = idxIn[((size_t)b * Q + q) * KNN + lane];
    int safe  = myIdx < 0 ? 0 : myIdx;
    float qx = qpos[((size_t)b * Q + q) * 2 + 0];
    float qy = qpos[((size_t)b * Q + q) * 2 + 1];
    const float2 prel = ((const float2*)ppos)[(size_t)b * P + safe];

    constexpr int DP = NKC1 * 32;       // padded row length actually used
    _Float16 rowv[DP];
    if (DF == 32) {
      const float4* f4 = (const float4*)(feats + ((size_t)b * P + safe) * DF);
      #pragma unroll
      for (int j = 0; j < 8; ++j) {
        float4 v = f4[j];
        rowv[j * 4 + 0] = (_Float16)v.x;
        rowv[j * 4 + 1] = (_Float16)v.y;
        rowv[j * 4 + 2] = (_Float16)v.z;
        rowv[j * 4 + 3] = (_Float16)v.w;
      }
    } else {
      const float2 f2 = ((const float2*)feats)[(size_t)b * P + safe];
      rowv[0] = (_Float16)f2.x;
      rowv[1] = (_Float16)f2.y;
    }
    rowv[DF]     = (_Float16)(prel.x - qx);
    rowv[DF + 1] = (_Float16)(prel.y - qy);
    #pragma unroll
    for (int d = DIN; d < DP; ++d) rowv[d] = (_Float16)0.0f;

    v8h* dst = (v8h*)&xw[lane * 64];
    #pragma unroll
    for (int j = 0; j < DP / 8; ++j) dst[j] = *(const v8h*)&rowv[j * 8];

    unsigned vm = (unsigned)__ballot(myIdx != -1);

    // ---- layer 1: H = gelu(X @ W1 + b1) ----
    #pragma unroll
    for (int mt = 0; mt < 2; ++mt) {
      const _Float16* arow = &xw[(mt * 16 + l15) * 64];
      HFrag a[NKC1];
      #pragma unroll
      for (int kc = 0; kc < NKC1; ++kc) {
        a[kc].h8[0] = *(const v8h*)(arow + kc * 32 + klo);
        a[kc].h8[1] = *(const v8h*)(arow + kc * 32 + klo + 8);
      }
      #pragma unroll
      for (int nt = 0; nt < 4; ++nt) {
        FFrag acc;
        #pragma unroll
        for (int i = 0; i < 8; ++i) acc.f[i] = 0.0f;
        #pragma unroll
        for (int kc = 0; kc < NKC1; ++kc)
          acc.v = __builtin_amdgcn_wmma_f32_16x16x32_f16(false, a[kc].v, false,
                     bf1[kc][nt].v, (short)0, acc.v, false, false);
        float bb = sB1v[nt * 16 + l15];
        #pragma unroll
        for (int i = 0; i < 8; ++i) {
          int m = mt * 16 + half8 + i;
          hw[m * 64 + nt * 16 + l15] = (_Float16)gelu_exact(acc.f[i] + bb);
        }
      }
    }

    // ---- layer 2: O = H @ W2 + b2, fused masked column-mean over 32 rows ----
    float colsum[4] = {0.f, 0.f, 0.f, 0.f};
    #pragma unroll
    for (int mt = 0; mt < 2; ++mt) {
      const _Float16* arow = &hw[(mt * 16 + l15) * 64];
      HFrag a[2];
      #pragma unroll
      for (int kc = 0; kc < 2; ++kc) {
        a[kc].h8[0] = *(const v8h*)(arow + kc * 32 + klo);
        a[kc].h8[1] = *(const v8h*)(arow + kc * 32 + klo + 8);
      }
      #pragma unroll
      for (int nt = 0; nt < 4; ++nt) {
        FFrag acc;
        #pragma unroll
        for (int i = 0; i < 8; ++i) acc.f[i] = 0.0f;
        #pragma unroll
        for (int kc = 0; kc < 2; ++kc)
          acc.v = __builtin_amdgcn_wmma_f32_16x16x32_f16(false, a[kc].v, false,
                     bf2[kc][nt].v, (short)0, acc.v, false, false);
        float bb = sB2v[nt * 16 + l15];
        float part = 0.f;
        #pragma unroll
        for (int i = 0; i < 8; ++i) {
          int m = mt * 16 + half8 + i;
          if ((vm >> m) & 1u) part += acc.f[i] + bb;
        }
        colsum[nt] += part;
      }
    }
    #pragma unroll
    for (int nt = 0; nt < 4; ++nt) colsum[nt] += __shfl_xor(colsum[nt], 16, 32);
    float inv = 1.0f / fmaxf((float)__popc(vm), 1.0f);

    if (lane < 16) {
      if (MODE == 0) {
        float* o = outP + ((size_t)b * Q + q) * 64;
        #pragma unroll
        for (int nt = 0; nt < 4; ++nt) o[nt * 16 + lane] = colsum[nt] * inv;
      } else {
        float* o = outP + ((size_t)b * Q + q) * 128 + colOff;
        #pragma unroll
        for (int nt = 0; nt < 4; ++nt) o[nt * 16 + lane] = colsum[nt] * inv;
      }
    }
  }
}

// ---------------------------------------------------------------------------
// Tiny MLPs / reductions (scalar; negligible FLOPs, deterministic order)
// ---------------------------------------------------------------------------
__global__ void penc_kernel(const float* __restrict__ gp,
                            const float* __restrict__ w1, const float* __restrict__ b1,
                            const float* __restrict__ w2, const float* __restrict__ b2,
                            float* __restrict__ rawctx) {
  __shared__ float sg[16];
  __shared__ float sh[64];
  int tid = threadIdx.x; // 64
  for (int b = 0; b < B; ++b) {
    if (tid < 16) sg[tid] = gp[b * 16 + tid];
    __syncthreads();
    float h = b1[tid];
    for (int i = 0; i < 16; ++i) h += sg[i] * w1[i * 64 + tid];
    sh[tid] = gelu_exact(h);
    __syncthreads();
    float o = b2[tid];
    for (int j = 0; j < 64; ++j) o += sh[j] * w2[j * 64 + tid];
    rawctx[b * 256 + tid] = o;
    __syncthreads();
  }
}

__global__ void cgeom_kernel(const float* __restrict__ gf,
                             const float* __restrict__ w1, const float* __restrict__ b1,
                             const float* __restrict__ w2, const float* __restrict__ b2,
                             float* __restrict__ rawctx) {
  __shared__ float sw1[128];
  __shared__ float sb1[64];
  __shared__ float sw2[64 * 64];
  __shared__ float sb2[64];
  __shared__ float sPart[4][64];
  int b = blockIdx.x, tid = threadIdx.x; // 256
  for (int i = tid; i < 64 * 64; i += 256) sw2[i] = w2[i];
  if (tid < 128) sw1[tid] = w1[tid];
  if (tid < 64) { sb1[tid] = b1[tid]; sb2[tid] = b2[tid]; }
  __syncthreads();
  int rg = tid >> 6, c = tid & 63;
  float acc = 0.f;
  for (int m = rg; m < MG; m += 4) {
    float2 f = ((const float2*)gf)[(size_t)b * MG + m];
    float o = 0.f;
    for (int j = 0; j < 64; ++j) {
      float h = gelu_exact(f.x * sw1[j] + f.y * sw1[64 + j] + sb1[j]);
      o += h * sw2[j * 64 + c];
    }
    acc += o;
  }
  sPart[rg][c] = acc;
  __syncthreads();
  if (tid < 64) {
    float t = sPart[0][tid] + sPart[1][tid] + sPart[2][tid] + sPart[3][tid];
    rawctx[b * 256 + 64 + tid] = t / (float)MG + sb2[tid];
  }
}

__global__ void reduceE_kernel(const float* __restrict__ meanE,
                               float* __restrict__ rawctx, int off) {
  int b = blockIdx.x, c = threadIdx.x; // 64
  float s = 0.f;
  for (int q = 0; q < MG; ++q) s += meanE[((size_t)b * MG + q) * 64 + c];
  rawctx[b * 256 + off + c] = s / (float)MG;
}

__global__ void proj_kernel(const float* __restrict__ rawctx,
                            const float* __restrict__ pw, const float* __restrict__ pb,
                            float* __restrict__ out) {
  __shared__ float sraw[B * 256];
  int tid = threadIdx.x; // 256
  for (int i = tid; i < B * 256; i += 256) sraw[i] = rawctx[i];
  __syncthreads();
  for (int b = 0; b < B; ++b) {
    float s = pb[tid];
    for (int i = 0; i < 256; ++i) s += sraw[b * 256 + i] * pw[i * 256 + tid];
    out[b * 256 + tid] = s;
  }
}

// ---------------------------------------------------------------------------
extern "C" void kernel_launch(void* const* d_in, const int* in_sizes, int n_in,
                              void* d_out, int out_size, void* d_ws, size_t ws_size,
                              hipStream_t stream) {
  (void)in_sizes; (void)n_in; (void)out_size; (void)ws_size;

  const float* global_params = (const float*)d_in[0];
  const float* geom_pos      = (const float*)d_in[1];
  const float* geom_feat     = (const float*)d_in[2];
  const float* in_pos        = (const float*)d_in[3];
  const float* in_feat       = (const float*)d_in[4];
  const float* gp_w1 = (const float*)d_in[5];
  const float* gp_b1 = (const float*)d_in[6];
  const float* gp_w2 = (const float*)d_in[7];
  const float* gp_b2 = (const float*)d_in[8];
  const float* gm_w1 = (const float*)d_in[9];
  const float* gm_b1 = (const float*)d_in[10];
  const float* gm_w2 = (const float*)d_in[11];
  const float* gm_b2 = (const float*)d_in[12];
  const float* phi_w1 = (const float*)d_in[13];
  const float* phi_b1 = (const float*)d_in[14];
  const float* phi_w2 = (const float*)d_in[15];
  const float* phi_b2 = (const float*)d_in[16];
  const float* psi_w1 = (const float*)d_in[17];
  const float* psi_b1 = (const float*)d_in[18];
  const float* psi_w2 = (const float*)d_in[19];
  const float* psi_b2 = (const float*)d_in[20];
  const float* proj_w = (const float*)d_in[21];
  const float* proj_b = (const float*)d_in[22];

  float* out    = (float*)d_out;
  float* rawctx = (float*)d_ws;                        // B*256 floats
  float* meanE  = rawctx + B * 256;                    // B*MG*64 floats (4 MB)
  int*   idxBuf = (int*)(meanE + (size_t)B * MG * 64); // max B*N*KNN ints (4 MB)
  float* aug    = out + B * 256;                       // (B,N,128)

  const float radii[2] = {0.1f, 0.2f};

  hipLaunchKernelGGL(penc_kernel, dim3(1), dim3(64), 0, stream,
                     global_params, gp_w1, gp_b1, gp_w2, gp_b2, rawctx);
  hipLaunchKernelGGL(cgeom_kernel, dim3(B), dim3(256), 0, stream,
                     geom_feat, gm_w1, gm_b1, gm_w2, gm_b2, rawctx);

  for (int s = 0; s < 2; ++s) {
    float r  = radii[s];
    float r2 = r * r;

    // input -> geometry (queries = geometry points, points = input points)
    hipLaunchKernelGGL(ballquery_kernel, dim3(B * ((MG + 63) / 64)), dim3(64), 0, stream,
                       geom_pos, in_pos, MG, N, r2, idxBuf);
    hipLaunchKernelGGL((mlp_gather_kernel<32, 34, 2, 0>), dim3(512), dim3(128), 0, stream,
                       in_feat, geom_pos, in_pos, idxBuf,
                       phi_w1 + (size_t)s * 34 * 64, phi_b1 + s * 64,
                       phi_w2 + (size_t)s * 64 * 64, phi_b2 + s * 64,
                       meanE, MG, N, 0);
    hipLaunchKernelGGL(reduceE_kernel, dim3(B), dim3(64), 0, stream,
                       meanE, rawctx, 128 + s * 64);

    // geometry -> input (queries = input points, points = geometry points)
    hipLaunchKernelGGL(ballquery_kernel, dim3(B * ((N + 63) / 64)), dim3(64), 0, stream,
                       in_pos, geom_pos, N, MG, r2, idxBuf);
    hipLaunchKernelGGL((mlp_gather_kernel<2, 4, 1, 1>), dim3(512), dim3(128), 0, stream,
                       geom_feat, in_pos, geom_pos, idxBuf,
                       psi_w1 + (size_t)s * 4 * 64, psi_b1 + s * 64,
                       psi_w2 + (size_t)s * 64 * 64, psi_b2 + s * 64,
                       aug, N, MG, s * 64);
  }

  hipLaunchKernelGGL(proj_kernel, dim3(1), dim3(256), 0, stream,
                     rawctx, proj_w, proj_b, out);
}
